// ContrastiveLoss_12128987644289
// MI455X (gfx1250) — matrix operand order repaired and checked by reading, not compiled, not run on hardware
//
#include <hip/hip_runtime.h>
#include <hip/hip_bf16.h>
#include <math.h>

typedef _Float16 v8h  __attribute__((ext_vector_type(8)));
typedef _Float16 v16h __attribute__((ext_vector_type(16)));
typedef float    v4f  __attribute__((ext_vector_type(4)));
typedef float    v8f  __attribute__((ext_vector_type(8)));

union H16 { v16h v; struct { v8h lo, hi; } s; };

#define BROWS   4096
#define NTOT    8192          // 2*BROWS
#define FDIM    256
#define INVT    10.0f         // 1/temperature
#define CSPLIT  4             // column-chunk split of the 512 N-tiles
#define TILES_TOTAL      (NTOT / 16)            // 512
#define TILES_PER_CHUNK  (TILES_TOTAL / CSPLIT) // 128
#define LDSROW  264           // 256 halves + 8 pad -> 528B stride, conflict-free

// ---------------------------------------------------------------------------
// Kernel 1: normalize rows, emit combined f16 matrix C=[n1;n2], and exact
// fp32 positive cosine similarity per row pair. One wave per row.
// ---------------------------------------------------------------------------
__global__ void __launch_bounds__(256)
normalize_kernel(const float* __restrict__ A, const float* __restrict__ B,
                 _Float16* __restrict__ C, float* __restrict__ pos)
{
    int wave = threadIdx.x >> 5;
    int lane = threadIdx.x & 31;
    int row  = blockIdx.x * 8 + wave;

    const float* ap = A + row * FDIM + lane * 8;
    const float* bp = B + row * FDIM + lane * 8;
    v4f a0 = *(const v4f*)ap, a1 = *(const v4f*)(ap + 4);
    v4f b0 = *(const v4f*)bp, b1 = *(const v4f*)(bp + 4);

    float ssa = 0.f, ssb = 0.f, pd = 0.f;
#pragma unroll
    for (int i = 0; i < 4; ++i) {
        ssa += a0[i] * a0[i] + a1[i] * a1[i];
        ssb += b0[i] * b0[i] + b1[i] * b1[i];
        pd  += a0[i] * b0[i] + a1[i] * b1[i];
    }
#pragma unroll
    for (int m = 16; m; m >>= 1) {
        ssa += __shfl_xor(ssa, m, 32);
        ssb += __shfl_xor(ssb, m, 32);
        pd  += __shfl_xor(pd,  m, 32);
    }
    float sa = 1.0f / fmaxf(sqrtf(ssa), 1e-7f);
    float sb = 1.0f / fmaxf(sqrtf(ssb), 1e-7f);
    if (lane == 0) pos[row] = pd * sa * sb;   // exact fp32 positive similarity

    v8h ha, hb;
#pragma unroll
    for (int i = 0; i < 4; ++i) {
        ha[i]     = (_Float16)(a0[i] * sa);
        ha[i + 4] = (_Float16)(a1[i] * sa);
        hb[i]     = (_Float16)(b0[i] * sb);
        hb[i + 4] = (_Float16)(b1[i] * sb);
    }
    *(v8h*)(C + (size_t)row * FDIM + lane * 8)           = ha;
    *(v8h*)(C + (size_t)(BROWS + row) * FDIM + lane * 8) = hb;
}

// ---------------------------------------------------------------------------
// Kernel 2: fused Gram-GEMM + fixed-max softmax row reduction.
// Wave -> 16-row strip (A resident in regs). Workgroup (8 waves) shares
// each 16-column tile via LDS. blockIdx.y splits the column space.
// Per row: r = sum_{j!=i} x_ij ; s = sum_{j!=i} exp(x_ij - 10), x = sim/T.
// ---------------------------------------------------------------------------
__global__ void __launch_bounds__(256)
simreduce_kernel(const _Float16* __restrict__ C,
                 float* __restrict__ rws, float* __restrict__ sws)
{
    __shared__ _Float16 lb[16 * LDSROW];

    int wave = threadIdx.x >> 5;
    int lane = threadIdx.x & 31;
    int rowbase = blockIdx.x * 128 + wave * 16;
    int cs = blockIdx.y;

    // Preload A strip (16x256 f16) per CDNA5 16-bit A-matrix layout:
    // lane<16: M=lane, K chunks {kk*32+0..7, kk*32+16..23}
    // lane>=16: M=lane-16, K chunks {kk*32+8..15, kk*32+24..31}
    H16 a[8];
    {
        const _Float16* ar = C + (size_t)(rowbase + (lane & 15)) * FDIM;
        int alo = (lane < 16) ? 0 : 8;
#pragma unroll
        for (int kk = 0; kk < 8; ++kk) {
            a[kk].s.lo = *(const v8h*)(ar + kk * 32 + alo);
            a[kk].s.hi = *(const v8h*)(ar + kk * 32 + alo + 16);
        }
    }

    float racc[8], sacc[8];
#pragma unroll
    for (int v = 0; v < 8; ++v) { racc[v] = 0.f; sacc[v] = 0.f; }

    int tr = threadIdx.x >> 4;   // staging: LDS row 0..15
    int tc = threadIdx.x & 15;   // staging: 16-half chunk 0..15
    // B per lane: K chunks {+0..7, +16..23} (lanes 0-15) / {+8..15, +24..31} (lanes 16-31).
    // lanes 0-15: N=lane, K base +0 ; lanes 16-31: N=lane-16, K base +8.
    const _Float16* bbase = &lb[(lane & 15) * LDSROW + ((lane < 16) ? 0 : 8)];

    for (int nt = 0; nt < TILES_PER_CHUNK; ++nt) {
        int colbase = (cs * TILES_PER_CHUNK + nt) * 16;

        const _Float16* src = C + (size_t)(colbase + tr) * FDIM + tc * 16;
        v8h p0 = *(const v8h*)src;
        v8h p1 = *(const v8h*)(src + 8);
        __syncthreads();                       // previous tile fully consumed
        *(v8h*)&lb[tr * LDSROW + tc * 16]     = p0;
        *(v8h*)&lb[tr * LDSROW + tc * 16 + 8] = p1;
        __syncthreads();                       // tile staged

        v8f cacc = {};
#pragma unroll
        for (int kk = 0; kk < 8; ++kk) {
            H16 b;
            b.s.lo = *(const v8h*)(bbase + kk * 32);
            b.s.hi = *(const v8h*)(bbase + kk * 32 + 16);
            cacc = __builtin_amdgcn_wmma_f32_16x16x32_f16(
                false, a[kk].v, false, b.v, (short)0, cacc, false, false);
        }

        // C/D layout: VGPR v, lanes 0-15 -> M=v; lanes 16-31 -> M=v+8; N=lane&15
        int mg0 = rowbase + ((lane >= 16) ? 8 : 0);
        int ng  = colbase + (lane & 15);
#pragma unroll
        for (int v = 0; v < 8; ++v) {
            float x  = cacc[v] * INVT;
            bool  dg = ((mg0 + v) == ng);       // exclude diagonal (-inf in ref)
            float e  = __expf(x - 10.0f);       // fixed max: x <= 10 guaranteed
            racc[v] += dg ? 0.0f : x;
            sacc[v] += dg ? 0.0f : e;
        }
    }

    // Merge the 16 lane-partials of each row (within each half-wave).
#pragma unroll
    for (int v = 0; v < 8; ++v) {
#pragma unroll
        for (int m = 8; m; m >>= 1) {
            racc[v] += __shfl_xor(racc[v], m, 32);
            sacc[v] += __shfl_xor(sacc[v], m, 32);
        }
    }
    if ((lane & 15) == 0) {
        int half = (lane >> 4) * 8;
#pragma unroll
        for (int v = 0; v < 8; ++v) {
            int row = rowbase + half + v;
            rws[row * CSPLIT + cs] = racc[v];
            sws[row * CSPLIT + cs] = sacc[v];
        }
    }
}

// ---------------------------------------------------------------------------
// Kernel 3: deterministic finalize. Sum per-row (r,s) over column chunks,
// form sum of off-diag log-probs, combine with positive term, write loss.
// ---------------------------------------------------------------------------
__global__ void __launch_bounds__(256)
finalize_kernel(const float* __restrict__ pos, const float* __restrict__ rws,
                const float* __restrict__ sws, float* __restrict__ out)
{
    __shared__ float red[256];
    float acc = 0.0f;
    for (int row = threadIdx.x; row < NTOT; row += 256) {
        float r = 0.f, s = 0.f;
#pragma unroll
        for (int c = 0; c < CSPLIT; ++c) {
            r += rws[row * CSPLIT + c];
            s += sws[row * CSPLIT + c];
        }
        // sum_{j!=i} log_prob_ij = r - (2B-1)*(10 + log s)
        acc += r - (float)(NTOT - 1) * (10.0f + __logf(s));
    }
    float pacc = 0.0f;
    for (int i = threadIdx.x; i < BROWS; i += 256) pacc += pos[i];

    float t = acc * (1.0f / (4.0f * (float)BROWS * (float)BROWS))
            - pacc * (INVT / (float)BROWS);
    red[threadIdx.x] = t;
    __syncthreads();
#pragma unroll
    for (int off = 128; off; off >>= 1) {
        if ((int)threadIdx.x < off) red[threadIdx.x] += red[threadIdx.x + off];
        __syncthreads();
    }
    if (threadIdx.x == 0) out[0] = red[0];
}

// ---------------------------------------------------------------------------
extern "C" void kernel_launch(void* const* d_in, const int* in_sizes, int n_in,
                              void* d_out, int out_size, void* d_ws, size_t ws_size,
                              hipStream_t stream)
{
    const float* A = (const float*)d_in[0];   // first_transformed  [4096,256] f32
    const float* B = (const float*)d_in[1];   // second_transformed [4096,256] f32

    char* ws = (char*)d_ws;
    _Float16* Cmat = (_Float16*)ws;                                  // 4 MB
    float* pos = (float*)(ws + (size_t)NTOT * FDIM * sizeof(_Float16)); // 16 KB
    float* rws = pos + BROWS;                                        // 128 KB
    float* sws = rws + (size_t)NTOT * CSPLIT;                        // 128 KB

    normalize_kernel<<<BROWS / 8, 256, 0, stream>>>(A, B, Cmat, pos);

    dim3 g2(NTOT / 128, CSPLIT);
    simreduce_kernel<<<g2, 256, 0, stream>>>(Cmat, rws, sws);

    finalize_kernel<<<1, 256, 0, stream>>>(pos, rws, sws, (float*)d_out);
}
